// NeRF_85315230368439
// MI455X (gfx1250) — matrix operand order repaired and checked by reading, not compiled, not run on hardware
//
#include <hip/hip_runtime.h>
#include <hip/hip_bf16.h>

typedef __attribute__((ext_vector_type(16))) _Float16 v16h;
typedef __attribute__((ext_vector_type(8)))  _Float16 v8h;
typedef __attribute__((ext_vector_type(8)))  float    v8f;

namespace {
constexpr int NFREQ   = 6;
constexpr int IN_DIM  = 39;    // 3 + 3*2*6
constexpr int K0P     = 64;    // input K padded to multiple of 32
constexpr int K0S     = 72;    // LDS row stride (halves) for K0-major rows (bank rotation)
constexpr int HDIM    = 256;
constexpr int HS      = 264;   // LDS row stride (halves) for H-major rows
constexpr int SSAMP   = 64;
constexpr int RPB     = 2;     // rays per block
constexpr int MT      = RPB * SSAMP;  // 128 rows per block
constexpr int THREADS = 256;   // 8 wave32

// LDS carve (bytes); all offsets 16B aligned
constexpr int OFF_W1  = 0;
constexpr int OFF_W0  = OFF_W1  + 256 * HS  * 2;   // W1: [256][HS] f16
constexpr int OFF_W2  = OFF_W0  + 256 * K0S * 2;   // W0: [256][K0S] f16
constexpr int OFF_H1  = OFF_W2  + 16  * HS  * 2;   // W2: [16][HS] f16
constexpr int OFF_XH2 = OFF_H1  + MT  * HS  * 2;   // H1: [128][HS] f16
constexpr int OFF_B0  = OFF_XH2 + MT  * HS  * 2;   // X ([128][K0S]) aliased with H2 ([128][HS])
constexpr int OFF_B1  = OFF_B0  + 256 * 4;
constexpr int OFF_B2  = OFF_B1  + 256 * 4;
constexpr int OFF_OUT = OFF_B2  + 16  * 4;
constexpr int SMEM_BYTES = OFF_OUT + MT * 4 * 4;   // 319808 B < 320KB WGP LDS
} // namespace

// ---- WMMA fragment loads from LDS (contiguous K = 16*hi + s, same order for A and B) ----------

// A (16x32 f16): lane row m = lane&15; halves = 16 contiguous K starting at 16*hi
__device__ __forceinline__ v16h load_a(const _Float16* __restrict__ rowk0, int hi) {
    v8h lo = *(const v8h*)(rowk0 + hi * 16);
    v8h hv = *(const v8h*)(rowk0 + hi * 16 + 8);
    v16h a;
#pragma unroll
    for (int i = 0; i < 8; ++i) { a[i] = lo[i]; a[8 + i] = hv[i]; }
    return a;
}

// B (32x16 f16): lane col n = lane&15; halves = 16 contiguous K starting at 16*hi.
// Weights stored transposed in LDS as [N][Kpad], so this is two contiguous b128 loads.
__device__ __forceinline__ v16h load_b(const _Float16* __restrict__ p) {
    v8h lo = *(const v8h*)(p);
    v8h hv = *(const v8h*)(p + 8);
    v16h b;
#pragma unroll
    for (int i = 0; i < 8; ++i) { b[i] = lo[i]; b[8 + i] = hv[i]; }
    return b;
}

__device__ __forceinline__ v8f wmma16(v16h a, v16h b, v8f c) {
    return __builtin_amdgcn_wmma_f32_16x16x32_f16(false, a, false, b, (short)0, c, false, false);
}

// One hidden layer: D[128,256] = relu(A[128,K] * W[K,256] + bias), f16 out to LDS.
// Each wave owns row-tile mt=wave; loops 16 N-tiles, 2 accumulators for ILP.
template<int KTILES, int AS, int WS, int OS>
__device__ __forceinline__ void gemm_relu(const _Float16* __restrict__ A,
                                          const _Float16* __restrict__ Wt,
                                          const float*    __restrict__ bias,
                                          _Float16*       __restrict__ O,
                                          int wave, int lane) {
    const int lane15 = lane & 15;
    const int hi     = lane >> 4;
    const _Float16* arow = A + (wave * 16 + lane15) * AS;
#pragma unroll 1
    for (int nt = 0; nt < 16; nt += 2) {
        v8f c0 = {}; v8f c1 = {};
        const _Float16* bp0 = Wt + (nt * 16 + lane15) * WS + hi * 16;
        const _Float16* bp1 = bp0 + 16 * WS;
#pragma unroll
        for (int kt = 0; kt < KTILES; ++kt) {
            const int k0 = kt * 32;
            v16h a  = load_a(arow + k0, hi);
            v16h b0 = load_b(bp0 + k0);
            v16h b1 = load_b(bp1 + k0);
            c0 = wmma16(a, b0, c0);
            c1 = wmma16(a, b1, c1);
        }
        const int   col0 = nt * 16 + lane15;
        const float bv0 = bias[col0], bv1 = bias[col0 + 16];
#pragma unroll
        for (int r = 0; r < 8; ++r) {
            const int row = wave * 16 + r + 8 * hi;   // C/D layout: VGPR r -> M = r + 8*hi
            O[row * OS + col0]      = (_Float16)fmaxf(c0[r] + bv0, 0.f);
            O[row * OS + col0 + 16] = (_Float16)fmaxf(c1[r] + bv1, 0.f);
        }
    }
}

__global__ __launch_bounds__(THREADS, 1)
void nerf_fused(const float* __restrict__ origins, const float* __restrict__ dirs,
                const float* __restrict__ trand,
                const float* __restrict__ W0, const float* __restrict__ b0,
                const float* __restrict__ W1, const float* __restrict__ b1,
                const float* __restrict__ W2, const float* __restrict__ b2,
                const int* __restrict__ nearp, const int* __restrict__ farp,
                float* __restrict__ out) {
    extern __shared__ char smem[];
    _Float16* sW1 = (_Float16*)(smem + OFF_W1);
    _Float16* sW0 = (_Float16*)(smem + OFF_W0);
    _Float16* sW2 = (_Float16*)(smem + OFF_W2);
    _Float16* sH1 = (_Float16*)(smem + OFF_H1);
    _Float16* sX  = (_Float16*)(smem + OFF_XH2);   // [128][K0S], aliased with...
    _Float16* sH2 = (_Float16*)(smem + OFF_XH2);   // [128][HS]
    float*    sB0 = (float*)(smem + OFF_B0);
    float*    sB1 = (float*)(smem + OFF_B1);
    float*    sB2 = (float*)(smem + OFF_B2);
    float*    sO  = (float*)(smem + OFF_OUT);      // [128][4] rgb,sigma

    const int tid  = threadIdx.x;
    const int wave = tid >> 5;
    const int lane = tid & 31;

    const float nearf = (float)nearp[0];
    const float farf  = (float)farp[0];
    const float delta = (farf - nearf) / (float)SSAMP;

    // ---- stage weights (global f32, coalesced) -> LDS f16 transposed [N][Kpad] ----
    for (int i = tid; i < 256 * 256; i += THREADS) {          // W1
        const int k = i >> 8, n = i & 255;
        sW1[n * HS + k] = (_Float16)W1[i];
    }
    for (int i = tid; i < K0P * 256; i += THREADS) {          // W0 (K padded 39->64 with zeros)
        const int k = i >> 8, n = i & 255;
        sW0[n * K0S + k] = (_Float16)((k < IN_DIM) ? W0[k * 256 + n] : 0.f);
    }
    for (int i = tid; i < 256 * 16; i += THREADS) {           // W2 (N padded 4->16 with zeros)
        const int k = i >> 4, n = i & 15;
        sW2[n * HS + k] = (_Float16)((n < 4) ? W2[k * 4 + n] : 0.f);
    }
    if (tid < 256) { sB0[tid] = b0[tid]; sB1[tid] = b1[tid]; }
    if (tid < 16)  { sB2[tid] = (tid < 4) ? b2[tid] : 0.f; }

    // ---- generate stratified samples + positional encoding -> sX [128][K0S] f16 ----
    if (tid < MT) {
        const int p = tid, lr = p >> 6, s = p & 63;
        const int gray = blockIdx.x * RPB + lr;
        const float tr = trand[gray * SSAMP + s];
        // linspace strata: z_rand = near + delta*(s + t_rand)
        const float z = nearf + delta * ((float)s + tr);
        float pt[3];
#pragma unroll
        for (int c = 0; c < 3; ++c)
            pt[c] = origins[gray * 3 + c] + dirs[gray * 3 + c] * z;
        _Float16* xr = sX + p * K0S;
#pragma unroll
        for (int c = 0; c < 3; ++c) xr[c] = (_Float16)pt[c];
        int idx = 3;
#pragma unroll
        for (int l = 0; l < NFREQ; ++l) {
            const float f = (float)(1 << l);
#pragma unroll
            for (int c = 0; c < 3; ++c) xr[idx + c]     = (_Float16)__sinf(pt[c] * f);
#pragma unroll
            for (int c = 0; c < 3; ++c) xr[idx + 3 + c] = (_Float16)__cosf(pt[c] * f);
            idx += 6;
        }
#pragma unroll
        for (int j = IN_DIM; j < K0P; ++j) xr[j] = (_Float16)0.f;
    }
    __syncthreads();

    // ---- layer 0: [128,64] x [64,256] -> H1, relu ----
    gemm_relu<2, K0S, K0S, HS>(sX, sW0, sB0, sH1, wave, lane);
    __syncthreads();
    // ---- layer 1: [128,256] x [256,256] -> H2 (aliased over sX), relu ----
    gemm_relu<8, HS, HS, HS>(sH1, sW1, sB1, sH2, wave, lane);
    __syncthreads();

    // ---- layer 2: [128,256] x [256,16] -> rgb(sigmoid)/sigma(relu) ----
    {
        const int lane15 = lane & 15, hi = lane >> 4;
        const _Float16* arow = sH2 + (wave * 16 + lane15) * HS;
        const _Float16* bp   = sW2 + lane15 * HS + hi * 16;
        v8f c = {};
#pragma unroll
        for (int kt = 0; kt < 8; ++kt) {
            const int k0 = kt * 32;
            c = wmma16(load_a(arow + k0, hi), load_b(bp + k0), c);
        }
        const int col = lane15;
        if (col < 4) {
            const float bv = sB2[col];
#pragma unroll
            for (int r = 0; r < 8; ++r) {
                const int row = wave * 16 + r + 8 * hi;
                float v = c[r] + bv;
                v = (col < 3) ? (1.f / (1.f + __expf(-v))) : fmaxf(v, 0.f);
                sO[row * 4 + col] = v;
            }
        }
    }
    __syncthreads();

    // ---- alpha compositing (seg_len == delta for uniform strata) ----
    if (tid < RPB) {
        const int gray = blockIdx.x * RPB + tid;
        const float* op = sO + tid * SSAMP * 4;
        float T = 1.f, r = 0.f, g = 0.f, bcol = 0.f;
        for (int s = 0; s < SSAMP; ++s) {
            const float sigma = op[s * 4 + 3];
            const float alpha = 1.f - __expf(-sigma * delta);
            const float w = alpha * T;
            r += w * op[s * 4 + 0];
            g += w * op[s * 4 + 1];
            bcol += w * op[s * 4 + 2];
            T *= (1.f - alpha + 1e-10f);
        }
        out[gray * 3 + 0] = r;
        out[gray * 3 + 1] = g;
        out[gray * 3 + 2] = bcol;
    }
}

extern "C" void kernel_launch(void* const* d_in, const int* in_sizes, int n_in,
                              void* d_out, int out_size, void* d_ws, size_t ws_size,
                              hipStream_t stream) {
    const float* origins = (const float*)d_in[0];
    const float* dirs    = (const float*)d_in[1];
    const float* trand   = (const float*)d_in[2];
    const float* W0      = (const float*)d_in[3];
    const float* b0      = (const float*)d_in[4];
    const float* W1      = (const float*)d_in[5];
    const float* b1      = (const float*)d_in[6];
    const float* W2      = (const float*)d_in[7];
    const float* b2      = (const float*)d_in[8];
    const int*   nearp   = (const int*)d_in[9];
    const int*   farp    = (const int*)d_in[10];
    float*       out     = (float*)d_out;

    const int Brays  = in_sizes[0] / 3;           // 16384
    const int nblock = Brays / RPB;               // 8192 blocks (one per WGP slot)
    nerf_fused<<<dim3(nblock), dim3(THREADS), SMEM_BYTES, stream>>>(
        origins, dirs, trand, W0, b0, W1, b1, W2, b2, nearp, farp, out);
}
